// QLSTM_65481071401569
// MI455X (gfx1250) — matrix operand order repaired and checked by reading, not compiled, not run on hardware
//
#include <hip/hip_runtime.h>
#include <hip/hip_fp16.h>

#define T_STEPS 512
#define BATCH   64
#define DIN     512
#define HID     512
#define DH      1024      // DIN + HID
#define NGATES  2048      // 4 * HID
#define NTILES  128       // NGATES / 16

typedef __attribute__((ext_vector_type(16))) _Float16 v16h;
typedef __attribute__((ext_vector_type(8)))  float    v8f;

// ---- A-fragment loaders for V_WMMA_F32_16X16X32_F16 ------------------------
// ISA 7.12.2, 16-bit A 16x32: lane(0-15)=row M, half=lane/16 adds +8 to K;
// VGPR v<4 holds K=2v,2v+1 ; VGPR v>=4 holds K=16+2(v-4),+1  (each +8*half).
__device__ __forceinline__ v16h load_a_f32(const float* __restrict__ src,
                                           int row_stride, int m, int kbase, int half) {
  v16h a;
  const float* p = src + (size_t)m * row_stride + kbase + half * 8;
#pragma unroll
  for (int j = 0; j < 8; ++j) {
    int d = (j < 4) ? (2 * j) : (2 * j + 8);   // elems 0-7 <- K 0..7, 8-15 <- K 16..23
    float2 v = *(const float2*)(p + d);
    a[2 * j]     = (_Float16)v.x;
    a[2 * j + 1] = (_Float16)v.y;
  }
  return a;
}

__device__ __forceinline__ v16h load_a_f16(const _Float16* __restrict__ src,
                                           int row_stride, int m, int kbase, int half) {
  v16h a;
  const _Float16* p = src + (size_t)m * row_stride + kbase + half * 8;
#pragma unroll
  for (int j = 0; j < 8; ++j) {
    int d = (j < 4) ? (2 * j) : (2 * j + 8);
    a[2 * j]     = p[d];
    a[2 * j + 1] = p[d + 1];
  }
  return a;
}

// ---- Weight pre-swizzle: B-fragment layout ---------------------------------
// B 32x16 (KxN) f16: lane%16 = N column; half=lane/16; element e<8 holds
// K=8*half+e, element e>=8 holds K=16+8*half+(e-8) (matches A loaders).
// Fragment for (ktile kt, ntile nt) stored contiguously: 32 lanes * 16 f16.
__global__ void lstm_pack_w(const float* __restrict__ Wf, const float* __restrict__ Wi,
                            const float* __restrict__ Wg, const float* __restrict__ Wo,
                            _Float16* __restrict__ wsB) {
  int idx  = blockIdx.x * blockDim.x + threadIdx.x;      // < DH*NGATES
  int e    = idx & 15;
  int lane = (idx >> 4) & 31;
  int tile = idx >> 9;
  int nt   = tile & (NTILES - 1);
  int kt   = tile >> 7;                                  // 0..31 over DH
  int K    = kt * 32 + e + ((e >> 3) << 3) + ((lane >> 4) << 3);
  int N    = nt * 16 + (lane & 15);
  int gate = N >> 9;
  int col  = N & 511;
  const float* W = (gate == 0) ? Wf : (gate == 1) ? Wi : (gate == 2) ? Wg : Wo;
  wsB[idx] = (_Float16)W[(size_t)K * HID + col];
}

// Pack the 4 bias vectors into one contiguous [2048] array (kills the
// exec-mask pointer-select ladder in the GEMM epilogue).
__global__ void lstm_pack_bias(const float* __restrict__ bf, const float* __restrict__ bi,
                               const float* __restrict__ bg, const float* __restrict__ bo,
                               float* __restrict__ ball) {
  int n = blockIdx.x * blockDim.x + threadIdx.x;         // < 2048
  int gate = n >> 9;
  int col  = n & 511;
  const float* b = (gate == 0) ? bf : (gate == 1) ? bi : (gate == 2) ? bg : bo;
  ball[n] = b[col];
}

__global__ void lstm_init(_Float16* __restrict__ h0, float* __restrict__ c0) {
  int i = blockIdx.x * blockDim.x + threadIdx.x;         // 32768 threads
  h0[i] = (_Float16)0.f;
  c0[i] = 0.f;
}

// ---- Phase A: XW[t*B + b, n] = ball[n] + sum_d x[t,b,d] * W[d,n] -----------
// Each wave: 1 M-tile x 8 N-tiles -> 8 WMMA per A-fragment load.
__global__ void __launch_bounds__(256) lstm_xw_gemm(
    const float* __restrict__ x, const _Float16* __restrict__ wsB,
    const float* __restrict__ ball, float* __restrict__ XW) {
  int lane = threadIdx.x & 31;
  int wave = blockIdx.x * 8 + (threadIdx.x >> 5);        // 32768 waves
  int mt   = wave >> 4;                                  // 0..2047 (M tiles)
  int ng   = wave & 15;                                  // 8 N-tiles per wave
  int half = lane >> 4;
  int m    = mt * 16 + (lane & 15);

  v8f z = {0.f, 0.f, 0.f, 0.f, 0.f, 0.f, 0.f, 0.f};
  v8f acc[8];
#pragma unroll
  for (int i = 0; i < 8; ++i) acc[i] = z;

  for (int kc = 0; kc < 16; ++kc) {                      // K = 512 = 16 * 32
    v16h a = load_a_f32(x, DIN, m, kc * 32, half);
#pragma unroll
    for (int i = 0; i < 8; ++i) {
      int nt = ng * 8 + i;
      const v16h* bp = (const v16h*)(wsB + (((size_t)(kc * NTILES + nt) * 32 + lane) << 4));
      acc[i] = __builtin_amdgcn_wmma_f32_16x16x32_f16(false, a, false, *bp,
                                                      (short)0, acc[i], false, false);
    }
  }
#pragma unroll
  for (int i = 0; i < 8; ++i) {
    int N  = (ng * 8 + i) * 16 + (lane & 15);
    float bv = ball[N];
#pragma unroll
    for (int r = 0; r < 8; ++r) {                        // C/D: M = r + 8*half
      int M = mt * 16 + r + 8 * half;
      XW[(size_t)M * NGATES + N] = acc[i][r] + bv;
    }
  }
}

// ---- Phase B: one recurrent step. Each wave: 1 M-tile x 1 col-tile over all
// four gates, so gating fuses in-register. 128 waves total. ------------------
__global__ void __launch_bounds__(256) lstm_step(
    const _Float16* __restrict__ wsB, const float* __restrict__ XWt,
    const _Float16* __restrict__ hin, _Float16* __restrict__ hout,
    float* __restrict__ c, float* __restrict__ outT) {
  int lane = threadIdx.x & 31;
  int wave = blockIdx.x * 8 + (threadIdx.x >> 5);        // 0..127
  int mt   = wave & 3;                                   // 64 rows -> 4 M tiles
  int ct   = wave >> 2;                                  // 0..31 col tiles
  int half = lane >> 4;
  int m    = mt * 16 + (lane & 15);

  v8f z = {0.f, 0.f, 0.f, 0.f, 0.f, 0.f, 0.f, 0.f};
  v8f acc[4];
#pragma unroll
  for (int g = 0; g < 4; ++g) acc[g] = z;

  // Prefetch first k-chunk's B fragments toward the WGP (L2-resident weights).
#pragma unroll
  for (int g = 0; g < 4; ++g) {
    int nt = g * 32 + ct;
    __builtin_prefetch(wsB + (((size_t)(16 * NTILES + nt) * 32 + lane) << 4), 0, 0);
  }

  for (int kc = 0; kc < 16; ++kc) {                      // K = HID = 512
    v16h a = load_a_f16(hin, HID, m, kc * 32, half);
    int kt = 16 + kc;                                    // recurrent rows of W
    if (kc + 1 < 16) {
#pragma unroll
      for (int g = 0; g < 4; ++g) {
        int nt = g * 32 + ct;
        __builtin_prefetch(wsB + (((size_t)((kt + 1) * NTILES + nt) * 32 + lane) << 4), 0, 0);
      }
    }
#pragma unroll
    for (int g = 0; g < 4; ++g) {
      int nt = g * 32 + ct;                              // same cols, 4 gates
      const v16h* bp = (const v16h*)(wsB + (((size_t)(kt * NTILES + nt) * 32 + lane) << 4));
      acc[g] = __builtin_amdgcn_wmma_f32_16x16x32_f16(false, a, false, *bp,
                                                      (short)0, acc[g], false, false);
    }
  }

  int col = ct * 16 + (lane & 15);
#pragma unroll
  for (int r = 0; r < 8; ++r) {
    int M = mt * 16 + r + 8 * half;
    const float* xwrow = XWt + (size_t)M * NGATES;
    float fv = 1.f / (1.f + __expf(-(acc[0][r] + xwrow[0 * 512 + col])));
    float iv = 1.f / (1.f + __expf(-(acc[1][r] + xwrow[1 * 512 + col])));
    float gv = tanhf(acc[2][r] + xwrow[2 * 512 + col]);
    float ov = 1.f / (1.f + __expf(-(acc[3][r] + xwrow[3 * 512 + col])));
    size_t ix = (size_t)M * HID + col;
    float cn = fv * c[ix] + iv * gv;
    c[ix] = cn;
    float hv = ov * tanhf(cn);
    outT[ix] = hv;                // f32 output for this timestep
    hout[ix] = (_Float16)hv;      // f16 state for next step's A-fragments
  }
}

__global__ void lstm_tail(float* __restrict__ out, const float* __restrict__ c) {
  int i = blockIdx.x * blockDim.x + threadIdx.x;         // 32768
  const size_t OUT_ELEMS = (size_t)T_STEPS * BATCH * HID;
  out[OUT_ELEMS + i] = out[(size_t)(T_STEPS - 1) * BATCH * HID + i];   // h_last
  out[OUT_ELEMS + (size_t)BATCH * HID + i] = c[i];                      // c_last
}

extern "C" void kernel_launch(void* const* d_in, const int* in_sizes, int n_in,
                              void* d_out, int out_size, void* d_ws, size_t ws_size,
                              hipStream_t stream) {
  (void)in_sizes; (void)n_in; (void)out_size; (void)ws_size;
  const float* x  = (const float*)d_in[0];
  const float* Wf = (const float*)d_in[1];
  const float* bf = (const float*)d_in[2];
  const float* Wi = (const float*)d_in[3];
  const float* bi = (const float*)d_in[4];
  const float* Wg = (const float*)d_in[5];
  const float* bg = (const float*)d_in[6];
  const float* Wo = (const float*)d_in[7];
  const float* bo = (const float*)d_in[8];
  float* out = (float*)d_out;
  char*  ws  = (char*)d_ws;

  // ws layout: [B-frags f16: 4MB][XW f32: 256MB][ball f32][h ping f16][h pong f16][c f32]
  _Float16* wsB  = (_Float16*)ws;
  float*    XW   = (float*)(ws + (size_t)4 * 1024 * 1024);
  float*    ball = (float*)(ws + (size_t)4 * 1024 * 1024 +
                            (size_t)T_STEPS * BATCH * NGATES * sizeof(float));
  _Float16* h0   = (_Float16*)(ball + NGATES);
  _Float16* h1   = h0 + (size_t)BATCH * HID;
  float*    c    = (float*)(h1 + (size_t)BATCH * HID);

  lstm_pack_w<<<(DH * NGATES) / 256, 256, 0, stream>>>(Wf, Wi, Wg, Wo, wsB);
  lstm_pack_bias<<<NGATES / 256, 256, 0, stream>>>(bf, bi, bg, bo, ball);
  lstm_init<<<(BATCH * HID) / 256, 256, 0, stream>>>(h0, c);
  lstm_xw_gemm<<<4096, 256, 0, stream>>>(x, wsB, ball, XW);

  for (int t = 0; t < T_STEPS; ++t) {
    const _Float16* hin = (t & 1) ? h1 : h0;
    _Float16*      hout = (t & 1) ? h0 : h1;
    lstm_step<<<16, 256, 0, stream>>>(wsB, XW + (size_t)t * BATCH * NGATES,
                                      hin, hout, c, out + (size_t)t * BATCH * HID);
  }
  lstm_tail<<<(BATCH * HID) / 256, 256, 0, stream>>>(out, c);
}
